// FineGrainedMixtureOfMLP_94489280665
// MI455X (gfx1250) — compile-verified, matrix operand order checked by reading, not run on hardware
//
#include <hip/hip_runtime.h>

#define TT 2048
#define HH 1024
#define II 512
#define EE 16
#define KK 4
#define NPAIR (TT * KK)

typedef __attribute__((ext_vector_type(16))) __bf16 v16bf;
typedef __attribute__((ext_vector_type(8))) float v8f;
typedef __attribute__((ext_vector_type(4))) unsigned int u32x4;
typedef __attribute__((ext_vector_type(8))) int i32x8;
typedef __attribute__((ext_vector_type(4))) int i32x4;

union BfPack {
    struct { uint4 lo, hi; } u;   // 32 bytes
    v16bf v;                      // 16 x bf16 = 32 bytes
};

__device__ __forceinline__ unsigned short f2bf(float f) {
    unsigned int u = __float_as_uint(f);
    u += 0x7fffu + ((u >> 16) & 1u);   // round-to-nearest-even
    return (unsigned short)(u >> 16);
}

__device__ __forceinline__ void load_pack(BfPack& d, const unsigned short* p, int chunk) {
    const uint4* q = (const uint4*)p;
    d.u.lo = q[chunk];
    d.u.hi = q[chunk + 2];
}

__device__ __forceinline__ void wait_tensor0() {
#if __has_builtin(__builtin_amdgcn_s_wait_tensorcnt)
    __builtin_amdgcn_s_wait_tensorcnt(0);
#else
    asm volatile("s_wait_tensorcnt 0x0" ::: "memory");
#endif
}

// ---------------------------------------------------------------------------
// 1) Routing: softmax -> top-4 -> renormalize (== softmax over selected logits)
// ---------------------------------------------------------------------------
__global__ void moe_route(const float* __restrict__ logits,
                          int* __restrict__ route_e, float* __restrict__ route_w) {
    int t = blockIdx.x * blockDim.x + threadIdx.x;
    if (t >= TT) return;
    float l[EE];
#pragma unroll
    for (int e = 0; e < EE; ++e) l[e] = logits[t * EE + e];
    unsigned int used = 0;
    int ids[KK]; float lv[KK];
#pragma unroll
    for (int k = 0; k < KK; ++k) {
        float best = -3.4e38f; int bi = 0;
#pragma unroll
        for (int e = 0; e < EE; ++e)
            if (!((used >> e) & 1u) && l[e] > best) { best = l[e]; bi = e; }
        used |= (1u << bi);
        ids[k] = bi; lv[k] = best;
    }
    float m = lv[0], s = 0.f, w[KK];
#pragma unroll
    for (int k = 0; k < KK; ++k) { w[k] = __expf(lv[k] - m); s += w[k]; }
    float inv = 1.0f / s;
#pragma unroll
    for (int k = 0; k < KK; ++k) {
        route_e[t * KK + k] = ids[k];
        route_w[t * KK + k] = w[k] * inv;
    }
}

// ---------------------------------------------------------------------------
// 2) Deterministic stable counting sort of (t,k) pairs by expert.
// ---------------------------------------------------------------------------
__global__ void moe_bucket(const int* __restrict__ route_e, const float* __restrict__ route_w,
                           int* __restrict__ sorted_pair, int* __restrict__ sorted_tok,
                           float* __restrict__ sorted_w, int* __restrict__ expert_base) {
    __shared__ int cnt[256][EE];
    __shared__ int eb[EE + 1];
    const int tid = threadIdx.x;
    const int per = NPAIR / 256;   // 32
    const int p0 = tid * per;
#pragma unroll
    for (int e = 0; e < EE; ++e) cnt[tid][e] = 0;
    for (int i = 0; i < per; ++i) cnt[tid][route_e[p0 + i]]++;
    __syncthreads();
    if (tid == 0) {
        int tot[EE];
        for (int e = 0; e < EE; ++e) {
            int run = 0;
            for (int t = 0; t < 256; ++t) { int c = cnt[t][e]; cnt[t][e] = run; run += c; }
            tot[e] = run;
        }
        int b = 0;
        for (int e = 0; e < EE; ++e) { eb[e] = b; b += tot[e]; }
        eb[EE] = b;
        for (int e = 0; e <= EE; ++e) expert_base[e] = eb[e];
    }
    __syncthreads();
    for (int i = 0; i < per; ++i) {
        int p = p0 + i;
        int e = route_e[p];
        int pos = eb[e] + cnt[tid][e]++;   // own LDS cell: race-free, stable
        sorted_pair[pos] = p;
        sorted_tok[pos]  = p >> 2;
        sorted_w[pos]    = route_w[p];
    }
}

// ---------------------------------------------------------------------------
// 3) f32 -> bf16 conversion (grid-stride)
// ---------------------------------------------------------------------------
__global__ void moe_cvt(const float* __restrict__ src, unsigned short* __restrict__ dst, int n) {
    int i = blockIdx.x * blockDim.x + threadIdx.x;
    int stride = gridDim.x * blockDim.x;
    for (; i < n; i += stride) dst[i] = f2bf(src[i]);
}

// ---------------------------------------------------------------------------
// 4) Up/Gate grouped GEMM + silu.
//    A strip (16 gathered token rows x H) staged into LDS by one TDM
//    gather-mode load (row indices = sorted token ids). B register
//    ping-pong double-buffered with STATIC indices (no scratch spill).
// ---------------------------------------------------------------------------
__global__ void __launch_bounds__(256)
moe_upgate(const unsigned short* __restrict__ xb,
           const unsigned short* __restrict__ wub,
           const unsigned short* __restrict__ wgb,
           const int* __restrict__ sorted_tok, const float* __restrict__ sorted_w,
           const int* __restrict__ expert_base,
           unsigned short* __restrict__ hb) {
    __shared__ unsigned short lds_a[16 * HH];   // 32 KB gathered A strip

    const int e    = blockIdx.y;
    const int seg0 = expert_base[e];
    const int n    = expert_base[e + 1] - seg0;
    const int row0 = blockIdx.x * 16;
    if (row0 >= n) return;                     // uniform across block

    const int lane  = threadIdx.x & 31;
    const int wv    = threadIdx.x >> 5;
    const int l16   = lane & 15;
    const int chunk = lane >> 4;               // A/B 16-bit operand layout selector
    const int colbase = wv * 64;

    // ---- stage gathered A strip -> LDS ----
#if __has_builtin(__builtin_amdgcn_tensor_load_to_lds)
    if (wv == 0) {
        // 16 row indices (clamped to valid sorted positions; tokens < 2048)
        unsigned int idx[16];
#pragma unroll
        for (int i = 0; i < 16; ++i) {
            int s = seg0 + row0 + i;
            if (s > NPAIR - 1) s = NPAIR - 1;
            idx[i] = (unsigned int)sorted_tok[s] & 0xFFFFu;
        }
        const unsigned long long gaddr = (unsigned long long)(uintptr_t)xb;
        const unsigned int ldsoff = (unsigned int)(uintptr_t)(&lds_a[0]);
        // group0: count=1 | gather_mode=1, 16-bit indices | lds | global | type=2
        u32x4 g0 = { 0x80000001u, ldsoff, (unsigned int)gaddr,
                     (unsigned int)((gaddr >> 32) & 0x01FFFFFFu) | 0x80000000u };
        // group1: data_size=2B; tensor HHxTT; tile_dim0=HH, tile_dim1=#idx=16; stride0=HH
        i32x8 g1 = { (int)(1u << 16),
                     (int)((unsigned)(HH & 0xFFFF) << 16),
                     (int)(((unsigned)TT & 0xFFFFu) << 16),
                     (int)((unsigned)HH << 16),
                     16, HH, 0, 0 };
        i32x4 g2 = { (int)(idx[0] | (idx[1] << 16)),  (int)(idx[2] | (idx[3] << 16)),
                     (int)(idx[4] | (idx[5] << 16)),  (int)(idx[6] | (idx[7] << 16)) };
        i32x4 g3 = { (int)(idx[8] | (idx[9] << 16)),  (int)(idx[10] | (idx[11] << 16)),
                     (int)(idx[12] | (idx[13] << 16)), (int)(idx[14] | (idx[15] << 16)) };
#if defined(__clang_major__) && __clang_major__ >= 23
        i32x8 z8 = { 0, 0, 0, 0, 0, 0, 0, 0 };
        __builtin_amdgcn_tensor_load_to_lds(g0, g1, g2, g3, z8, 0);
#else
        __builtin_amdgcn_tensor_load_to_lds(g0, g1, g2, g3, 0);
#endif
        wait_tensor0();
    }
#else
    {   // fallback: cooperative gathered copy
        for (int i = threadIdx.x; i < 16 * (HH / 8); i += 256) {
            int r = i / (HH / 8), c = i % (HH / 8);
            int s = seg0 + row0 + r; if (s > NPAIR - 1) s = NPAIR - 1;
            ((uint4*)lds_a)[i] =
                ((const uint4*)(xb + (size_t)sorted_tok[s] * HH))[c];
        }
    }
#endif
    __syncthreads();

    const unsigned short* bub = wub + ((size_t)e * II + colbase + l16) * HH;
    const unsigned short* bgb = wgb + ((size_t)e * II + colbase + l16) * HH;
    const unsigned short* arow = lds_a + (size_t)l16 * HH;

    v8f accU[4], accG[4];
    const v8f zero = {0.f, 0.f, 0.f, 0.f, 0.f, 0.f, 0.f, 0.f};
#pragma unroll
    for (int j = 0; j < 4; ++j) { accU[j] = zero; accG[j] = zero; }

    auto loadB = [&](BfPack (&u)[4], BfPack (&g)[4], int kb) {
#pragma unroll
        for (int j = 0; j < 4; ++j) {
            load_pack(u[j], bub + (size_t)j * 16 * HH + kb, chunk);
            load_pack(g[j], bgb + (size_t)j * 16 * HH + kb, chunk);
        }
    };
    auto loadA = [&](BfPack& a, int kb) {
        const uint4* lp = (const uint4*)(arow + kb);
        a.u.lo = lp[chunk];
        a.u.hi = lp[chunk + 2];
    };
    auto mmac = [&](BfPack& a, BfPack (&u)[4], BfPack (&g)[4]) {
#pragma unroll
        for (int j = 0; j < 4; ++j) {
            accU[j] = __builtin_amdgcn_wmma_f32_16x16x32_bf16(
                false, a.v, false, u[j].v, (short)0, accU[j], false, false);
            accG[j] = __builtin_amdgcn_wmma_f32_16x16x32_bf16(
                false, a.v, false, g[j].v, (short)0, accG[j], false, false);
        }
    };

    BfPack a, u0[4], g0[4], u1[4], g1[4];
    loadB(u0, g0, 0);
    for (int kb = 0; kb < HH; kb += 64) {       // unroll-by-2 ping-pong, static indices
        loadB(u1, g1, kb + 32);
        loadA(a, kb);
        mmac(a, u0, g0);
        if (kb + 64 < HH) loadB(u0, g0, kb + 64);
        loadA(a, kb + 32);
        mmac(a, u1, g1);
    }

#pragma unroll
    for (int r = 0; r < 8; ++r) {
        const int m   = r + chunk * 8;         // C/D layout: M = r (+8 for hi lanes)
        const int row = row0 + m;
        if (row < n) {
            const int s2 = seg0 + row;
            const float wgt = sorted_w[s2];
            unsigned short* hrow = hb + (size_t)s2 * II;
#pragma unroll
            for (int j = 0; j < 4; ++j) {
                float g = accG[j][r] * wgt;
                float u = accU[j][r] * wgt;
                float hv = (g / (1.0f + __expf(-g))) * u;   // silu(g)*u
                hrow[colbase + j * 16 + l16] = f2bf(hv);
            }
        }
    }
}

// ---------------------------------------------------------------------------
// 5) Down grouped GEMM. Contiguous 16-row h strip staged via TDM 2D tile
//    load; B register ping-pong double-buffered with static indices.
// ---------------------------------------------------------------------------
__global__ void __launch_bounds__(256)
moe_down(const unsigned short* __restrict__ hb,
         const unsigned short* __restrict__ wdb,
         const int* __restrict__ sorted_pair, const float* __restrict__ sorted_w,
         const int* __restrict__ expert_base,
         float* __restrict__ part) {
    __shared__ unsigned short lds_a[16 * II];   // 16 KB A strip

    const int e    = blockIdx.y;
    const int seg0 = expert_base[e];
    const int n    = expert_base[e + 1] - seg0;
    const int row0 = blockIdx.x * 16;
    if (row0 >= n) return;

    const int lane  = threadIdx.x & 31;
    const int wv    = threadIdx.x >> 5;
    const int l16   = lane & 15;
    const int chunk = lane >> 4;
    const int colbase = wv * 128;

    // ---- stage A strip: rows [seg0+row0, +16) x II bf16 -> LDS ----
#if __has_builtin(__builtin_amdgcn_tensor_load_to_lds)
    if (wv == 0) {
        const unsigned long long gaddr =
            (unsigned long long)(uintptr_t)(hb + (size_t)(seg0 + row0) * II);
        const unsigned int ldsoff = (unsigned int)(uintptr_t)(&lds_a[0]);
        u32x4 g0 = { 1u, ldsoff, (unsigned int)gaddr,
                     (unsigned int)((gaddr >> 32) & 0x01FFFFFFu) | 0x80000000u };
        i32x8 g1 = { (int)(1u << 16),
                     (int)((unsigned)(II & 0xFFFF) << 16),
                     (int)(((unsigned)NPAIR & 0xFFFFu) << 16),
                     (int)((unsigned)II << 16),
                     16, II, 0, 0 };
        i32x4 z4 = { 0, 0, 0, 0 };
#if defined(__clang_major__) && __clang_major__ >= 23
        i32x8 z8 = { 0, 0, 0, 0, 0, 0, 0, 0 };
        __builtin_amdgcn_tensor_load_to_lds(g0, g1, z4, z4, z8, 0);
#else
        __builtin_amdgcn_tensor_load_to_lds(g0, g1, z4, z4, 0);
#endif
        wait_tensor0();
    }
#else
    {   // fallback: cooperative contiguous copy
        const uint4* src = (const uint4*)(hb + (size_t)(seg0 + row0) * II);
        uint4* dst = (uint4*)lds_a;
        for (int i = threadIdx.x; i < 16 * II / 8; i += 256) dst[i] = src[i];
    }
#endif
    __syncthreads();

    const unsigned short* bb = wdb + ((size_t)e * HH + colbase + l16) * II;
    const unsigned short* arow = lds_a + (size_t)l16 * II;

    v8f acc[8];
    const v8f zero = {0.f, 0.f, 0.f, 0.f, 0.f, 0.f, 0.f, 0.f};
#pragma unroll
    for (int j = 0; j < 8; ++j) acc[j] = zero;

    auto loadB = [&](BfPack (&b)[8], int kb) {
#pragma unroll
        for (int j = 0; j < 8; ++j)
            load_pack(b[j], bb + (size_t)j * 16 * II + kb, chunk);
    };
    auto loadA = [&](BfPack& a, int kb) {
        const uint4* lp = (const uint4*)(arow + kb);
        a.u.lo = lp[chunk];
        a.u.hi = lp[chunk + 2];
    };
    auto mmac = [&](BfPack& a, BfPack (&b)[8]) {
#pragma unroll
        for (int j = 0; j < 8; ++j)
            acc[j] = __builtin_amdgcn_wmma_f32_16x16x32_bf16(
                false, a.v, false, b[j].v, (short)0, acc[j], false, false);
    };

    BfPack a, b0[8], b1[8];
    loadB(b0, 0);
    for (int kb = 0; kb < II; kb += 64) {       // unroll-by-2 ping-pong, static indices
        loadB(b1, kb + 32);
        loadA(a, kb);
        mmac(a, b0);
        if (kb + 64 < II) loadB(b0, kb + 64);
        loadA(a, kb + 32);
        mmac(a, b1);
    }

#pragma unroll
    for (int r = 0; r < 8; ++r) {
        const int m   = r + chunk * 8;
        const int row = row0 + m;
        if (row < n) {
            const int s2 = seg0 + row;
            const float wgt = sorted_w[s2];
            const int p = sorted_pair[s2];
            float* orow = part + (size_t)p * HH;
#pragma unroll
            for (int j = 0; j < 8; ++j)
                orow[colbase + j * 16 + l16] = acc[j][r] * wgt;
        }
    }
}

// ---------------------------------------------------------------------------
// 6) Fixed-order reduction over k -> deterministic output
// ---------------------------------------------------------------------------
__global__ void moe_reduce(const float* __restrict__ part, float* __restrict__ out) {
    int idx = blockIdx.x * blockDim.x + threadIdx.x;
    if (idx >= TT * HH) return;
    int t = idx / HH;
    int h = idx - t * HH;
    const float* p = part + (size_t)t * KK * HH + h;
    out[idx] = ((p[0] + p[HH]) + p[2 * HH]) + p[3 * HH];
}

// ---------------------------------------------------------------------------
extern "C" void kernel_launch(void* const* d_in, const int* in_sizes, int n_in,
                              void* d_out, int out_size, void* d_ws, size_t ws_size,
                              hipStream_t stream) {
    const float* x      = (const float*)d_in[0];
    const float* logits = (const float*)d_in[1];
    const float* w_up   = (const float*)d_in[2];
    const float* w_gate = (const float*)d_in[3];
    const float* w_down = (const float*)d_in[4];
    float* out = (float*)d_out;

    char* ws = (char*)d_ws;
    size_t off = 0;
    auto alloc = [&](size_t bytes) -> void* {
        off = (off + 255) & ~(size_t)255;
        void* p = ws + off;
        off += bytes;
        return p;
    };

    int*   route_e     = (int*)alloc((size_t)NPAIR * 4);
    float* route_w     = (float*)alloc((size_t)NPAIR * 4);
    int*   sorted_pair = (int*)alloc((size_t)NPAIR * 4);
    int*   sorted_tok  = (int*)alloc((size_t)NPAIR * 4);
    float* sorted_w    = (float*)alloc((size_t)NPAIR * 4);
    int*   expert_base = (int*)alloc((EE + 1) * 4);
    unsigned short* xb  = (unsigned short*)alloc((size_t)TT * HH * 2);
    unsigned short* wub = (unsigned short*)alloc((size_t)EE * II * HH * 2);
    unsigned short* wgb = (unsigned short*)alloc((size_t)EE * II * HH * 2);
    unsigned short* wdb = (unsigned short*)alloc((size_t)EE * HH * II * 2);
    unsigned short* hb  = (unsigned short*)alloc((size_t)NPAIR * II * 2);
    float* part         = (float*)alloc((size_t)NPAIR * HH * 4);

    moe_route<<<(TT + 255) / 256, 256, 0, stream>>>(logits, route_e, route_w);
    moe_bucket<<<1, 256, 0, stream>>>(route_e, route_w, sorted_pair, sorted_tok,
                                      sorted_w, expert_base);
    moe_cvt<<<2048, 256, 0, stream>>>(x, xb, TT * HH);
    moe_cvt<<<4096, 256, 0, stream>>>(w_up,   wub, EE * II * HH);
    moe_cvt<<<4096, 256, 0, stream>>>(w_gate, wgb, EE * II * HH);
    moe_cvt<<<4096, 256, 0, stream>>>(w_down, wdb, EE * HH * II);
    moe_upgate<<<dim3(TT / 16, EE), 256, 0, stream>>>(xb, wub, wgb, sorted_tok,
                                                      sorted_w, expert_base, hb);
    moe_down<<<dim3(TT / 16, EE), 256, 0, stream>>>(hb, wdb, sorted_pair, sorted_w,
                                                    expert_base, part);
    moe_reduce<<<(TT * HH + 255) / 256, 256, 0, stream>>>(part, out);
}
